// Local_SelfAttenion_Layer_83133386982121
// MI455X (gfx1250) — matrix operand order repaired and checked by reading, not compiled, not run on hardware
//
#include <hip/hip_runtime.h>
#include <hip/hip_bf16.h>

typedef __bf16 bf16;
typedef __attribute__((ext_vector_type(16))) __bf16 v16bf;
typedef __attribute__((ext_vector_type(8)))  __bf16 v8bf;
typedef __attribute__((ext_vector_type(8)))  float  v8f;

#define SEQ    1024
#define BATCH  8
#define DIN    512
#define EMB    1024
#define NHEAD  16
#define HDIM   64
#define NTOK   (SEQ*BATCH)   /* 8192 */
#define WIN    128
#define NKT    17            /* key tiles per 16-query tile */
#define KPAD   288           /* 18*16 padded key columns   */

__device__ inline v8f vzero8() {
  v8f v;
#pragma unroll
  for (int i = 0; i < 8; ++i) v[i] = 0.f;
  return v;
}
__device__ inline v16bf vzero16b() {
  v16bf v;
#pragma unroll
  for (int i = 0; i < 16; ++i) v[i] = (bf16)0.f;
  return v;
}
__device__ inline v16bf cat8(v8bf lo, v8bf hi) {
  return __builtin_shufflevector(lo, hi, 0, 1, 2, 3, 4, 5, 6, 7,
                                 8, 9, 10, 11, 12, 13, 14, 15);
}
__device__ inline v8f wmma_bf(v16bf a, v16bf b, v8f c) {
  return __builtin_amdgcn_wmma_f32_16x16x32_bf16(false, a, false, b, (short)0, c,
                                                 false, false);
}

// ---- fragment loaders (ISA 7.12.2 layouts, wave32) -------------------------
// A 16x32 bf16: lane<16 -> K 0..15 ; lane>=16 -> K 16..31 (contiguous, matches B)
__device__ inline v16bf frag_a_bf16(const bf16* A, int lda, int row, int k0, int lane) {
  const bf16* p = A + (size_t)row * lda + k0 + ((lane >> 4) << 4);
  return cat8(*(const v8bf*)p, *(const v8bf*)(p + 8));
}
__device__ inline v16bf frag_a_f32(const float* A, int lda, int row, int k0, int lane) {
  const float* p = A + (size_t)row * lda + k0 + ((lane >> 4) << 4);
  const v8f f0 = *(const v8f*)p;
  const v8f f1 = *(const v8f*)(p + 8);
  v16bf r;
#pragma unroll
  for (int i = 0; i < 8; ++i) { r[i] = (bf16)f0[i]; r[i + 8] = (bf16)f1[i]; }
  return r;
}
// B 32x16 bf16: lane n holds column n&15, K contiguous (lane<16: 0..15, else 16..31)
__device__ inline v16bf frag_b(const bf16* Bm, int ldb, int col, int k0, int lane) {
  const bf16* p = Bm + (size_t)col * ldb + k0 + ((lane >> 4) << 4);
  return cat8(*(const v8bf*)p, *(const v8bf*)(p + 8));
}

// ---- epilogues (one call per 16x16 tile; r strides rows from a single base) -
struct EpiStoreF32 {
  float* C; const float* bias; int ldc;
  __device__ void store8(int m, int n, v8f v) const {
    float* p = C + (size_t)m * ldc + n;
    const float bb = bias ? bias[n] : 0.f;
#pragma unroll
    for (int r = 0; r < 8; ++r) p[(size_t)r * ldc] = v[r] + bb;
  }
};
// scatter into head-major q[H,B,L,64], k[H,B,L,64], vT[H,B,64,L] (bf16);
// batch stride = 128 KB -> fits the 24-bit VMEM immediate offset
struct EpiQKV {
  bf16 *q, *k, *vt;
  __device__ void store8(int m, int n, v8f v) const {
    // m is a multiple of 8 -> token l = m>>3 fixed, batch index b = r
    const int which = n >> 10, e = n & 1023;
    const int hh = e >> 6, hd = e & 63;
    const int l = m >> 3;
    if (which == 2) {
      bf16* p = vt + ((size_t)(hh * BATCH) * HDIM + hd) * SEQ + l;
#pragma unroll
      for (int r = 0; r < 8; ++r) p[(size_t)r * (HDIM * SEQ)] = (bf16)v[r];
    } else {
      bf16* p = (which ? k : q) + ((size_t)(hh * BATCH) * SEQ + l) * HDIM + hd;
#pragma unroll
      for (int r = 0; r < 8; ++r) p[(size_t)r * (SEQ * HDIM)] = (bf16)v[r];
    }
  }
};

// ---- WMMA GEMM: C[M,N] = A[M,K] * B[N,K]^T  (TN, both K-contiguous) --------
template <bool AF32, class Epi>
__global__ __launch_bounds__(256) void gemm_tn(const void* __restrict__ Ap,
                                               const bf16* __restrict__ Bm,
                                               int K, Epi epi) {
  const int lane = threadIdx.x & 31;
  const int wave = threadIdx.x >> 5;
  const int lr = lane & 15, hi = lane >> 4;
  const int m0 = blockIdx.y * 128 + (wave & 3) * 32;
  const int n0 = blockIdx.x * 128 + (wave >> 2) * 64;
  v8f acc[2][4];
#pragma unroll
  for (int i = 0; i < 2; ++i)
#pragma unroll
    for (int j = 0; j < 4; ++j) acc[i][j] = vzero8();

  for (int k0 = 0; k0 < K; k0 += 32) {
    v16bf a[2], b[4];
#pragma unroll
    for (int i = 0; i < 2; ++i) {
      const int row = m0 + 16 * i + lr;
      if constexpr (AF32) a[i] = frag_a_f32((const float*)Ap, K, row, k0, lane);
      else                a[i] = frag_a_bf16((const bf16*)Ap, K, row, k0, lane);
    }
#pragma unroll
    for (int j = 0; j < 4; ++j) b[j] = frag_b(Bm, K, n0 + 16 * j + lr, k0, lane);
#pragma unroll
    for (int i = 0; i < 2; ++i)
#pragma unroll
      for (int j = 0; j < 4; ++j) acc[i][j] = wmma_bf(a[i], b[j], acc[i][j]);
  }
#pragma unroll
  for (int i = 0; i < 2; ++i)
#pragma unroll
    for (int j = 0; j < 4; ++j)
      epi.store8(m0 + 16 * i + hi * 8, n0 + 16 * j + lr, acc[i][j]);
}

// ---- banded attention ------------------------------------------------------
__global__ __launch_bounds__(128) void k_attn(const bf16* __restrict__ qh,
                                              const bf16* __restrict__ kh,
                                              const bf16* __restrict__ vtp,
                                              const int* __restrict__ slv,
                                              bf16* __restrict__ ctxb,
                                              float* __restrict__ att) {
  __shared__ __align__(16) float s_mean[16 * KPAD];
  __shared__ __align__(16) bf16  s_w[4][16 * KPAD];
  const int tid = threadIdx.x;
  const int lane = tid & 31, wave = tid >> 5;
  const int lr = lane & 15, hi = lane >> 4;
  const int q0 = blockIdx.x * 16;
  const int b  = blockIdx.y;
  const int ktg = q0 - WIN;          // key of tile0/col0
  const int slb = slv[b];

  for (int i = tid; i < 16 * KPAD; i += 128) s_mean[i] = 0.f;
  for (int i = lane; i < 16 * 16; i += 32)   // zero padding tile of own region
    s_w[wave][(i >> 4) * KPAD + 272 + (i & 15)] = (bf16)0.f;
  __syncthreads();

  for (int hseg = 0; hseg < 4; ++hseg) {
    const int head = hseg * 4 + wave;
    const bf16* Q  = qh  + (size_t)(head * BATCH + b) * SEQ * HDIM;
    const bf16* Km = kh  + (size_t)(head * BATCH + b) * SEQ * HDIM;
    const bf16* Vt = vtp + (size_t)(head * BATCH + b) * HDIM * SEQ;

    const v16bf aq0 = frag_a_bf16(Q, HDIM, q0 + lr, 0, lane);
    const v16bf aq1 = frag_a_bf16(Q, HDIM, q0 + lr, 32, lane);

    // scores S[16q, 272k] in 17 C-fragments
    v8f sc[NKT];
#pragma unroll
    for (int t = 0; t < NKT; ++t) {
      const int kt = ktg + t * 16;
      v8f s = vzero8();
      if (kt >= 0 && kt < SEQ) {
        const v16bf b0 = frag_b(Km, HDIM, kt + lr, 0, lane);
        const v16bf b1 = frag_b(Km, HDIM, kt + lr, 32, lane);
        s = wmma_bf(aq0, b0, s);
        s = wmma_bf(aq1, b1, s);
      }
      sc[t] = s;
    }
    // scale + mask (C layout: row = hi*8+j, col = tile*16+lr)
#pragma unroll
    for (int t = 0; t < NKT; ++t) {
      const int kcol = ktg + t * 16 + lr;
#pragma unroll
      for (int j = 0; j < 8; ++j) {
        const int qrow = q0 + hi * 8 + j;
        const bool ok = (kcol >= 0) && (kcol < slb) &&
                        (kcol >= qrow - WIN) && (kcol <= qrow + WIN);
        sc[t][j] = ok ? sc[t][j] * 0.125f : -1e30f;
      }
    }
    // softmax across 272 keys: reduce over 17 tiles + 16 lanes of half-wave
    float rinv[8], rmax[8];
#pragma unroll
    for (int j = 0; j < 8; ++j) {
      float m = sc[0][j];
#pragma unroll
      for (int t = 1; t < NKT; ++t) m = fmaxf(m, sc[t][j]);
      for (int off = 8; off; off >>= 1) m = fmaxf(m, __shfl_xor(m, off));
      rmax[j] = m;
    }
#pragma unroll
    for (int j = 0; j < 8; ++j) {
      float s = 0.f;
#pragma unroll
      for (int t = 0; t < NKT; ++t) {
        const float e = __expf(sc[t][j] - rmax[j]);
        sc[t][j] = e;
        s += e;
      }
      for (int off = 8; off; off >>= 1) s += __shfl_xor(s, off);
      rinv[j] = (rmax[j] > -1e29f && s > 0.f) ? 1.f / s : 0.f;  // masked row -> 0
    }
    // write w to LDS (bf16, for PV re-fragmentation) and head-mean accumulation
#pragma unroll
    for (int t = 0; t < NKT; ++t) {
#pragma unroll
      for (int j = 0; j < 8; ++j) {
        const int row = hi * 8 + j, col = t * 16 + lr;
        const float w = sc[t][j] * rinv[j];
        s_w[wave][row * KPAD + col] = (bf16)w;
        atomicAdd(&s_mean[row * KPAD + col], w * (1.0f / NHEAD));
      }
    }
    __syncthreads();
    // ctx[16,64] = W[16,288] * V[288,64]  (9 k-steps of 32)
    v8f ctx[4];
#pragma unroll
    for (int nt = 0; nt < 4; ++nt) ctx[nt] = vzero8();
    for (int ks = 0; ks < 9; ++ks) {
      const int k0 = ks * 32;
      const bf16* wp = &s_w[wave][lr * KPAD + k0 + (hi << 4)];
      const v16bf aw = cat8(*(const v8bf*)wp, *(const v8bf*)(wp + 8));
      const int kk = ktg + k0 + (hi << 4);
      const bool in = (kk >= 0) && (kk + 16 <= SEQ);
#pragma unroll
      for (int nt = 0; nt < 4; ++nt) {
        v16bf bv;
        if (in) {
          const bf16* vp = Vt + (size_t)(nt * 16 + lr) * SEQ + kk;
          bv = cat8(*(const v8bf*)vp, *(const v8bf*)(vp + 8));
        } else {
          bv = vzero16b();
        }
        ctx[nt] = wmma_bf(aw, bv, ctx[nt]);
      }
    }
    // store ctx as token-major bf16 [t, e]; j stride = BATCH*EMB*2 = 16 KB imm
#pragma unroll
    for (int nt = 0; nt < 4; ++nt) {
      bf16* p = ctxb + (size_t)((q0 + hi * 8) * BATCH + b) * EMB +
                head * HDIM + nt * 16 + lr;
#pragma unroll
      for (int j = 0; j < 8; ++j)
        p[(size_t)j * (BATCH * EMB)] = (bf16)ctx[nt][j];
    }
    __syncthreads();
  }
  // head-averaged attention map -> att[L,B,L] window band
  for (int i = tid; i < 16 * 272; i += 128) {
    const int r = i / 272, c = i % 272;
    const int kc = ktg + c;
    if (kc >= 0 && kc < SEQ)
      att[((size_t)((q0 + r) * BATCH + b)) * SEQ + kc] = s_mean[r * KPAD + c];
  }
}

// ---- residual + LayerNorm -> bf16 ------------------------------------------
__global__ __launch_bounds__(256) void k_ln(const float* __restrict__ h,
                                            const float* __restrict__ ao,
                                            const float* __restrict__ g,
                                            const float* __restrict__ be,
                                            bf16* __restrict__ out) {
  const int t = blockIdx.x, tid = threadIdx.x;
  __shared__ float red[256];
  float v[4], s = 0.f;
#pragma unroll
  for (int i = 0; i < 4; ++i) {
    const int e = tid + i * 256;
    v[i] = h[(size_t)t * EMB + e] + ao[(size_t)t * EMB + e];
    s += v[i];
  }
  red[tid] = s; __syncthreads();
  for (int o = 128; o > 0; o >>= 1) { if (tid < o) red[tid] += red[tid + o]; __syncthreads(); }
  const float mu = red[0] * (1.0f / EMB);
  __syncthreads();
  float s2 = 0.f;
#pragma unroll
  for (int i = 0; i < 4; ++i) { const float d = v[i] - mu; s2 += d * d; }
  red[tid] = s2; __syncthreads();
  for (int o = 128; o > 0; o >>= 1) { if (tid < o) red[tid] += red[tid + o]; __syncthreads(); }
  const float rstd = rsqrtf(red[0] * (1.0f / EMB) + 1e-5f);
#pragma unroll
  for (int i = 0; i < 4; ++i) {
    const int e = tid + i * 256;
    out[(size_t)t * EMB + e] = (bf16)((v[i] - mu) * rstd * g[e] + be[e]);
  }
}

// ---- small utilities -------------------------------------------------------
__global__ void k_f2b(const float* __restrict__ in, bf16* __restrict__ out, int n) {
  const int i = blockIdx.x * 256 + threadIdx.x;
  if (i < n) out[i] = (bf16)in[i];
}
__global__ void k_zero(float* __restrict__ p, int n) {
  const int i = blockIdx.x * 256 + threadIdx.x;
  if (i < n) p[i] = 0.f;
}

extern "C" void kernel_launch(void* const* d_in, const int* in_sizes, int n_in,
                              void* d_out, int out_size, void* d_ws, size_t ws_size,
                              hipStream_t stream) {
  (void)in_sizes; (void)n_in; (void)out_size; (void)ws_size;
  const float* x    = (const float*)d_in[0];
  const int*   sl   = (const int*)d_in[1];
  const float* W_in = (const float*)d_in[2];
  const float* b_in = (const float*)d_in[3];
  const float* Wqkv = (const float*)d_in[4];
  const float* Wao  = (const float*)d_in[5];
  const float* lng  = (const float*)d_in[6];
  const float* lnb  = (const float*)d_in[7];
  const float* Wout = (const float*)d_in[8];
  const float* bout = (const float*)d_in[9];

  const size_t MB = 1ull << 20;
  char* ws = (char*)d_ws;
  float* h    = (float*)(ws + 0);          // 32 MB  h (f32, kept for residual)
  bf16*  ctxb = (bf16*)(ws + 32 * MB);     // 16 MB  attention context (bf16)
  float* ao   = (float*)(ws + 48 * MB);    // 32 MB  attn out-proj (f32)
  bf16*  qh   = (bf16*)(ws + 80 * MB);     // 16 MB  Q [H,B,L,64]
  bf16*  kh   = (bf16*)(ws + 96 * MB);     // 16 MB  K [H,B,L,64]
  bf16*  vt   = (bf16*)(ws + 112 * MB);    // 16 MB  V^T [H,B,64,L]
  bf16*  rb   = (bf16*)(ws + 80 * MB);     // reuse qh after attention (LN out)
  bf16*  wb_in  = (bf16*)(ws + 128 * MB);  // 1 MB
  bf16*  wb_qkv = (bf16*)(ws + 130 * MB);  // 6 MB
  bf16*  wb_ao  = (bf16*)(ws + 137 * MB);  // 2 MB
  bf16*  wb_out = (bf16*)(ws + 140 * MB);  // 1 MB

  // weight conversion to bf16
  k_f2b<<<(EMB * DIN + 255) / 256, 256, 0, stream>>>(W_in, wb_in, EMB * DIN);
  k_f2b<<<(3 * EMB * EMB + 255) / 256, 256, 0, stream>>>(Wqkv, wb_qkv, 3 * EMB * EMB);
  k_f2b<<<(EMB * EMB + 255) / 256, 256, 0, stream>>>(Wao, wb_ao, EMB * EMB);
  k_f2b<<<(DIN * EMB + 255) / 256, 256, 0, stream>>>(Wout, wb_out, DIN * EMB);

  // zero the attention-map output (band writes happen later)
  float* att = (float*)d_out + (size_t)NTOK * DIN;
  k_zero<<<(NTOK * SEQ + 255) / 256, 256, 0, stream>>>(att, NTOK * SEQ);

  // h = x * W_in^T + b_in
  gemm_tn<true><<<dim3(EMB / 128, NTOK / 128), 256, 0, stream>>>(
      (const void*)x, wb_in, DIN, EpiStoreF32{h, b_in, EMB});
  // qkv = h * Wqkv^T, scattered per-head (V transposed)
  gemm_tn<true><<<dim3(3 * EMB / 128, NTOK / 128), 256, 0, stream>>>(
      (const void*)h, wb_qkv, EMB, EpiQKV{qh, kh, vt});
  // banded attention + head-mean attention map
  k_attn<<<dim3(SEQ / 16, BATCH), 128, 0, stream>>>(qh, kh, vt, sl, ctxb, att);
  // attn_out = ctx * Wao^T
  gemm_tn<false><<<dim3(EMB / 128, NTOK / 128), 256, 0, stream>>>(
      (const void*)ctxb, wb_ao, EMB, EpiStoreF32{ao, nullptr, EMB});
  // r = LN(h + attn_out) -> bf16
  k_ln<<<NTOK, 256, 0, stream>>>(h, ao, lng, lnb, rb);
  // y = r * W_out^T + b_out -> d_out
  gemm_tn<false><<<dim3(DIN / 128, NTOK / 128), 256, 0, stream>>>(
      (const void*)rb, wb_out, EMB, EpiStoreF32{(float*)d_out, bout, DIN});
}
